// FrontierLayerVN_42279658062116
// MI455X (gfx1250) — matrix operand order repaired and checked by reading, not compiled, not run on hardware
//
#include <hip/hip_runtime.h>
#include <hip/hip_bf16.h>

// -------------------- types --------------------
typedef __attribute__((ext_vector_type(16))) __bf16 v16bf;
typedef __attribute__((ext_vector_type(8)))  __bf16 v8bf;
typedef __attribute__((ext_vector_type(8)))  float  v8f;

// Explicit address spaces: ds_* for LDS, global_* for memory.
typedef __attribute__((address_space(3))) float        ldsf;  // LDS float
typedef __attribute__((address_space(1))) const float  gcf;   // global const f32
typedef __attribute__((address_space(1))) const __bf16 gbf;   // global const bf16
typedef __attribute__((address_space(1))) const v8bf   gv8bf; // global const bf16x8

constexpr int   LDC   = 321;  // cat (16 x 320) leading dim (pad: 64-bank LDS)
constexpr int   LDSX  = 257;  // s   (16 x 256) leading dim
constexpr int   LDV   = 65;   // vec (16 x 64)  leading dim
constexpr float SLOPE = 0.01f;
constexpr float EPSV  = 1e-6f;

// bf16 weight cache layout inside d_ws (element offsets); shapes fixed by ref.
constexpr int WOFF_A1WV1 = 0;
constexpr int WOFF_A1WV2 = 4096;
constexpr int WOFF_A1WD  = 8192;
constexpr int WOFF_A2WV1 = 12288;
constexpr int WOFF_N1WV1 = 16384;
constexpr int WOFF_N1WV2 = 20480;
constexpr int WOFF_N1WD  = 24576;
constexpr int WOFF_N2WV1 = 28672;
constexpr int WOFF_N2WV2 = 32768;
constexpr int WOFF_A1WS  = 36864;
constexpr int WOFF_N1WS  = 118784;
constexpr int WOFF_N2WS  = 200704;
constexpr int WOFF_A1WG  = 282624;
constexpr int WOFF_N1WG  = 299008;
constexpr int WOFF_N2WG  = 315392;
constexpr int WTOTAL     = 331776;

struct Smem {
  float cat[16 * LDC];     // cols 0..63 = |vi|, cols 64..319 = scalar features
  float v [3][16 * LDV];   // vector features (x,y,z planes)
  float vi[3][16 * LDV];   // hidden vector features / temps
  float s [16 * LDSX];     // scalar GEMM output
  float wgt[16];           // per-node softmax weight (net kernel)
};

// -------------------- WMMA fragment helpers --------------------
// A: 16x32 bf16, lane L holds row M=L&15; lanes>=16 take K offset +16
// (contiguous K, matching B). Converted from fp32 LDS.
__device__ inline v16bf load_A_frag(const ldsf* A, int lda, int k0, int lane) {
  const int m  = lane & 15;
  const int kb = k0 + (lane & 16);
  const ldsf* p = A + m * lda + kb;
  v16bf a;
#pragma unroll
  for (int e = 0; e < 8; ++e) a[e]     = (__bf16)p[e];
#pragma unroll
  for (int e = 0; e < 8; ++e) a[e + 8] = (__bf16)p[8 + e];
  return a;
}

// B = W^T, W bf16 row-major (NC x K): lane holds col N=L&15 (row of W),
// 16 contiguous bf16 (32B) -> two 16B vector loads, no converts.
__device__ inline v16bf load_Bt_frag(gbf* W, int ldw, int n0, int k0, int lane) {
  gbf* p = W + (n0 + (lane & 15)) * ldw + k0 + (lane & 16);
  gv8bf* p8 = (gv8bf*)p;
  v8bf lo = p8[0], hi = p8[1];
  return __builtin_shufflevector(lo, hi, 0, 1, 2, 3, 4, 5, 6, 7,
                                 8, 9, 10, 11, 12, 13, 14, 15);
}

// C/D f32 16x16: lane holds col N=L&15, VGPR r -> row r (+8 for lanes>=16).
__device__ inline void store_C_frag(ldsf* C, int ldc, int n0, v8f c, int lane) {
  const int n  = n0 + (lane & 15);
  const int mb = (lane & 16) >> 1;
#pragma unroll
  for (int r = 0; r < 8; ++r) C[(mb + r) * ldc + n] = c[r];
}

// C(16 x NC) = A(16 x 32*KS, LDS fp32) * W(NC x 32*KS, bf16)^T.
// A fragments preloaded once into registers and reused across all n-tiles.
template <int KS>
__device__ inline void gemm16(const ldsf* A, int lda, gbf* W, int ldw,
                              int NC, ldsf* C, int ldc, int lane) {
  v16bf afr[KS];
#pragma unroll
  for (int ks = 0; ks < KS; ++ks) afr[ks] = load_A_frag(A, lda, ks * 32, lane);
  for (int n0 = 0; n0 < NC; n0 += 16) {
    v8f acc = {};
#pragma unroll
    for (int ks = 0; ks < KS; ++ks) {
      v16bf b = load_Bt_frag(W, ldw, n0, ks * 32, lane);
      acc = __builtin_amdgcn_wmma_f32_16x16x32_bf16(false, afr[ks], false, b,
                                                    (short)0, acc, false, false);
    }
    store_C_frag(C, ldc, n0, acc, lane);
  }
}

// Three-plane channel-mix GEMM sharing the B matrix:
// C[d](16x64) = A[d](16x64) @ W(64x64)^T   for d = x,y,z.
// B fragments are loaded once per n-tile and reused for all 3 planes
// (6 WMMAs per 2 B-fragment fetches).
__device__ inline void gemm16_vn3(ldsf* const* A, int lda, gbf* W,
                                  ldsf* const* C, int ldc, int lane) {
  v16bf afr[3][2];
#pragma unroll
  for (int d = 0; d < 3; ++d)
#pragma unroll
    for (int ks = 0; ks < 2; ++ks)
      afr[d][ks] = load_A_frag(A[d], lda, ks * 32, lane);
  for (int n0 = 0; n0 < 64; n0 += 16) {
    v16bf b0 = load_Bt_frag(W, 64, n0, 0, lane);
    v16bf b1 = load_Bt_frag(W, 64, n0, 32, lane);
#pragma unroll
    for (int d = 0; d < 3; ++d) {
      v8f acc = {};
      acc = __builtin_amdgcn_wmma_f32_16x16x32_bf16(false, afr[d][0], false, b0,
                                                    (short)0, acc, false, false);
      acc = __builtin_amdgcn_wmma_f32_16x16x32_bf16(false, afr[d][1], false, b1,
                                                    (short)0, acc, false, false);
      store_C_frag(C[d], ldc, n0, acc, lane);
    }
  }
}

// -------------------- LDS views --------------------
struct SmemV {
  ldsf* cat;
  ldsf* v[3];
  ldsf* vi[3];
  ldsf* s;
  ldsf* wgt;
};
__device__ inline SmemV make_view(Smem* sm) {
  SmemV o;
  o.cat   = (ldsf*)sm->cat;
  o.v[0]  = (ldsf*)sm->v[0];  o.v[1]  = (ldsf*)sm->v[1];  o.v[2]  = (ldsf*)sm->v[2];
  o.vi[0] = (ldsf*)sm->vi[0]; o.vi[1] = (ldsf*)sm->vi[1]; o.vi[2] = (ldsf*)sm->vi[2];
  o.s     = (ldsf*)sm->s;
  o.wgt   = (ldsf*)sm->wgt;
  return o;
}

// -------------------- GVLinear (shared by all stages) --------------------
__device__ void gv_linear(const SmemV& sm, gbf* Wv1, gbf* Wv2,
                          gbf* Ws, gbf* Wg, gcf* bg, int lane) {
  __syncthreads();
  gemm16_vn3(sm.v, LDV, Wv1, sm.vi, LDV, lane);         // vi = Wv1 @ v
  __syncthreads();
  for (int i = lane; i < 16 * 64; i += 32) {            // vnorm -> cat[:,0:64]
    int m = i >> 6, c = i & 63;
    float x = sm.vi[0][m * LDV + c];
    float y = sm.vi[1][m * LDV + c];
    float z = sm.vi[2][m * LDV + c];
    sm.cat[m * LDC + c] = sqrtf(x * x + y * y + z * z);
  }
  __syncthreads();
  gemm16<10>(sm.cat, LDC, Ws, 320, 256, sm.s, LDSX, lane);  // out_sca
  gemm16_vn3(sm.vi, LDV, Wv2, sm.v, LDV, lane);             // out_vec = Wv2 @ vi
  __syncthreads();
  // gate GEMM (s @ Wg^T, 256->64) fused with sigmoid + vector gating
  {
    v16bf afr[8];
#pragma unroll
    for (int ks = 0; ks < 8; ++ks) afr[ks] = load_A_frag(sm.s, LDSX, ks * 32, lane);
    for (int n0 = 0; n0 < 64; n0 += 16) {
      v8f acc = {};
#pragma unroll
      for (int ks = 0; ks < 8; ++ks) {
        v16bf b = load_Bt_frag(Wg, 256, n0, ks * 32, lane);
        acc = __builtin_amdgcn_wmma_f32_16x16x32_bf16(false, afr[ks], false, b,
                                                      (short)0, acc, false, false);
      }
      const int n  = n0 + (lane & 15);
      const int mb = (lane & 16) >> 1;
      const float bgv = bg[n];
#pragma unroll
      for (int r = 0; r < 8; ++r) {
        float gate = 1.0f / (1.0f + __expf(-(acc[r] + bgv)));
        int m = mb + r;
        sm.v[0][m * LDV + n] *= gate;
        sm.v[1][m * LDV + n] *= gate;
        sm.v[2][m * LDV + n] *= gate;
      }
    }
  }
  __syncthreads();
}

// LeakyReLU(s), VNLeakyReLU(v; Wd), then cat[:,64:320] = s  (perceptron tail)
__device__ void gv_act(const SmemV& sm, gbf* Wd, int lane) {
  for (int i = lane; i < 16 * 256; i += 32) {
    int m = i >> 8, c = i & 255;
    float x = sm.s[m * LDSX + c];
    sm.s[m * LDSX + c] = x >= 0.f ? x : SLOPE * x;
  }
  __syncthreads();
  gemm16_vn3(sm.v, LDV, Wd, sm.vi, LDV, lane);          // d = Wd @ v
  __syncthreads();
  for (int i = lane; i < 16 * 64; i += 32) {
    int m = i >> 6, c = i & 63;
    float vx = sm.v[0][m*LDV+c], vy = sm.v[1][m*LDV+c], vz = sm.v[2][m*LDV+c];
    float dx = sm.vi[0][m*LDV+c], dy = sm.vi[1][m*LDV+c], dz = sm.vi[2][m*LDV+c];
    float dot = vx*dx + vy*dy + vz*dz;
    if (dot < 0.f) {
      float t = dot / (dx*dx + dy*dy + dz*dz + EPSV);
      sm.v[0][m*LDV+c] = SLOPE*vx + (1.f-SLOPE)*(vx - t*dx);
      sm.v[1][m*LDV+c] = SLOPE*vy + (1.f-SLOPE)*(vy - t*dy);
      sm.v[2][m*LDV+c] = SLOPE*vz + (1.f-SLOPE)*(vz - t*dz);
    }
  }
  __syncthreads();
  for (int i = lane; i < 16 * 256; i += 32) {           // next stage scalar in
    int m = i >> 8, c = i & 255;
    sm.cat[m * LDC + 64 + c] = sm.s[m * LDSX + c];
  }
  __syncthreads();
}

__device__ void load_inputs(const SmemV& sm, gcf* h_sca, gcf* h_vec,
                            gcf* te, int base, int N, int lane) {
  for (int i = lane; i < 16 * 256; i += 32) {
    int m = i >> 8, c = i & 255;
    int node = base + m; if (node >= N) node = N - 1;
    sm.cat[m * LDC + 64 + c] = h_sca[node * 256 + c] + te[c];
  }
  for (int i = lane; i < 16 * 64; i += 32) {
    int m = i >> 6, c = i & 63;
    int node = base + m; if (node >= N) node = N - 1;
    gcf* p = h_vec + (node * 64 + c) * 3;
    sm.v[0][m * LDV + c] = p[0];
    sm.v[1][m * LDV + c] = p[1];
    sm.v[2][m * LDV + c] = p[2];
  }
  __syncthreads();
}

// -------------------- kernels --------------------
__global__ void cvt_kernel(const float* __restrict__ src, __bf16* __restrict__ dst,
                           int n) {
  int i = blockIdx.x * blockDim.x + threadIdx.x;
  if (i < n) dst[i] = (__bf16)src[i];
}

__global__ __launch_bounds__(32) void att_branch_kernel(
    const float* __restrict__ h_sca_, const float* __restrict__ h_vec_,
    const float* __restrict__ time_embed_, const int* __restrict__ t_ptr,
    const __bf16* __restrict__ wbf_, const float* a1bg_, const float* a2Ws_,
    float* __restrict__ att_out, int N) {
  __shared__ Smem smraw;
  SmemV sm = make_view(&smraw);
  gcf* h_sca = (gcf*)h_sca_;  gcf* h_vec = (gcf*)h_vec_;
  gcf* temb  = (gcf*)time_embed_;
  gcf* a1bg  = (gcf*)a1bg_;   gcf* a2Ws  = (gcf*)a2Ws_;
  gbf* wbf   = (gbf*)wbf_;

  const int lane = threadIdx.x;
  const int base = blockIdx.x * 16;
  load_inputs(sm, h_sca, h_vec, temb + (*t_ptr) * 256, base, N, lane);

  gv_linear(sm, wbf + WOFF_A1WV1, wbf + WOFF_A1WV2, wbf + WOFF_A1WS,
            wbf + WOFF_A1WG, a1bg, lane);
  gv_act(sm, wbf + WOFF_A1WD, lane);

  // att2: vi = a2Wv1 @ v ; att = [ |vi| , s ] . a2Ws[0,:]
  gemm16_vn3(sm.v, LDV, wbf + WOFF_A2WV1, sm.vi, LDV, lane);
  __syncthreads();
  for (int i = lane; i < 16 * 64; i += 32) {
    int m = i >> 6, c = i & 63;
    float x = sm.vi[0][m*LDV+c], y = sm.vi[1][m*LDV+c], z = sm.vi[2][m*LDV+c];
    sm.cat[m * LDC + c] = sqrtf(x * x + y * y + z * z);
  }
  __syncthreads();
  const int m = lane & 15, half = lane >> 4;
  float acc = 0.f;
  for (int k = half * 160; k < half * 160 + 160; ++k)
    acc += sm.cat[m * LDC + k] * a2Ws[k];
  acc += __shfl_xor(acc, 16);
  if (lane < 16 && base + lane < N) att_out[base + lane] = acc;
}

__device__ inline void atomicMaxF(float* addr, float val) {
  unsigned int* ua = (unsigned int*)addr;
  unsigned int old = *ua, assumed;
  do {
    assumed = old;
    if (__uint_as_float(assumed) >= val) break;
    old = atomicCAS(ua, assumed, __float_as_uint(val));
  } while (old != assumed);
}

__global__ void init_kernel(float* out, int out_n, float* mseg, float* zseg, int B) {
  int i = blockIdx.x * blockDim.x + threadIdx.x;
  if (i < out_n) out[i] = 0.f;
  if (i < B) { mseg[i] = -3.402823e38f; zseg[i] = 0.f; }
}

__global__ void segmax_kernel(const float* att, const int* bid, float* mseg, int N) {
  int i = blockIdx.x * blockDim.x + threadIdx.x;
  if (i < N) atomicMaxF(&mseg[bid[i]], att[i]);
}

__global__ void segsum_kernel(const float* att, const int* bid,
                              const float* mseg, float* zseg, int N) {
  int i = blockIdx.x * blockDim.x + threadIdx.x;
  if (i < N) atomicAdd(&zseg[bid[i]], __expf(att[i] - mseg[bid[i]]));
}

__global__ __launch_bounds__(32) void net_branch_kernel(
    const float* __restrict__ h_sca_, const float* __restrict__ h_vec_,
    const float* __restrict__ pos_, const float* __restrict__ time_embed_,
    const int* __restrict__ t_ptr,
    const __bf16* __restrict__ wbf_, const float* n1bg_, const float* n2bg_,
    const float* __restrict__ att, const int* __restrict__ bid,
    const float* __restrict__ mseg, const float* __restrict__ zseg,
    float* __restrict__ out_feat, float* __restrict__ out_vec,
    float* __restrict__ out_pos, int N) {
  __shared__ Smem smraw;
  SmemV sm = make_view(&smraw);
  gcf* h_sca = (gcf*)h_sca_;  gcf* h_vec = (gcf*)h_vec_;
  gcf* pos   = (gcf*)pos_;    gcf* temb  = (gcf*)time_embed_;
  gcf* n1bg  = (gcf*)n1bg_;   gcf* n2bg  = (gcf*)n2bg_;
  gbf* wbf   = (gbf*)wbf_;

  const int lane = threadIdx.x;
  const int base = blockIdx.x * 16;
  load_inputs(sm, h_sca, h_vec, temb + (*t_ptr) * 256, base, N, lane);

  gv_linear(sm, wbf + WOFF_N1WV1, wbf + WOFF_N1WV2, wbf + WOFF_N1WS,
            wbf + WOFF_N1WG, n1bg, lane);
  gv_act(sm, wbf + WOFF_N1WD, lane);
  gv_linear(sm, wbf + WOFF_N2WV1, wbf + WOFF_N2WV2, wbf + WOFF_N2WS,
            wbf + WOFF_N2WG, n2bg, lane);
  // now: fs = sm.s (16x256), fv = sm.v (3 x 16x64)

  if (lane < 16) {
    int node = base + lane;
    float w = 0.f;
    if (node < N) {
      int b = bid[node];
      w = __expf(att[node] - mseg[b]) / zseg[b];
    }
    sm.wgt[lane] = w;
  }
  __syncthreads();

  for (int i = lane; i < 16 * 256; i += 32) {
    int m = i >> 8, c = i & 255;
    int node = base + m;
    if (node < N)
      atomicAdd(&out_feat[bid[node] * 256 + c], sm.wgt[m] * sm.s[m * LDSX + c]);
  }
  for (int i = lane; i < 16 * 64; i += 32) {
    int m = i >> 6, c = i & 63;
    int node = base + m;
    if (node < N) {
      int b = bid[node];
      float w = sm.wgt[m];
      atomicAdd(&out_vec[(b * 64 + c) * 3 + 0], w * sm.v[0][m * LDV + c]);
      atomicAdd(&out_vec[(b * 64 + c) * 3 + 1], w * sm.v[1][m * LDV + c]);
      atomicAdd(&out_vec[(b * 64 + c) * 3 + 2], w * sm.v[2][m * LDV + c]);
    }
  }
  for (int i = lane; i < 48; i += 32) {
    int m = i / 3, d = i % 3;
    int node = base + m;
    if (node < N)
      atomicAdd(&out_pos[bid[node] * 3 + d], sm.wgt[m] * pos[node * 3 + d]);
  }
}

// -------------------- host entry --------------------
extern "C" void kernel_launch(void* const* d_in, const int* in_sizes, int n_in,
                              void* d_out, int out_size, void* d_ws, size_t ws_size,
                              hipStream_t stream) {
  const float* h_sca = (const float*)d_in[0];
  const float* h_vec = (const float*)d_in[1];
  const float* pos   = (const float*)d_in[2];
  const float* temb  = (const float*)d_in[3];
  const float* a1bg  = (const float*)d_in[8];
  const float* a2Ws  = (const float*)d_in[12];
  const float* n1bg  = (const float*)d_in[19];
  const float* n2bg  = (const float*)d_in[25];
  const int*   t_ptr = (const int*)d_in[26];
  const int*   bid   = (const int*)d_in[27];

  const int N = in_sizes[0] / 256;        // 120000
  const int B = out_size / 451;           // 256 + 64*3 + 3 = 451 floats/segment

  float* att  = (float*)d_ws;             // N floats
  float* mseg = att + N;                  // B floats
  float* zseg = mseg + B;                 // B floats
  size_t wboff = (((size_t)(N + 2 * B) * 4) + 255) & ~(size_t)255;
  __bf16* wbf  = (__bf16*)((char*)d_ws + wboff);   // WTOTAL bf16 weights

  float* out_feat = (float*)d_out;        // B x 256
  float* out_vec  = out_feat + B * 256;   // B x 64 x 3
  float* out_pos  = out_vec + B * 192;    // B x 3

  // ---- convert all GEMM weights to bf16 once per call ----
  struct { int src_idx; int off; int n; } cv[15] = {
      { 4, WOFF_A1WV1,  4096}, { 5, WOFF_A1WV2,  4096}, { 9, WOFF_A1WD,  4096},
      {10, WOFF_A2WV1,  4096}, {15, WOFF_N1WV1,  4096}, {16, WOFF_N1WV2, 4096},
      {20, WOFF_N1WD,   4096}, {21, WOFF_N2WV1,  4096}, {22, WOFF_N2WV2, 4096},
      { 6, WOFF_A1WS,  81920}, {17, WOFF_N1WS,  81920}, {23, WOFF_N2WS, 81920},
      { 7, WOFF_A1WG,  16384}, {18, WOFF_N1WG,  16384}, {24, WOFF_N2WG, 16384}};
  for (int i = 0; i < 15; ++i)
    cvt_kernel<<<(cv[i].n + 255) / 256, 256, 0, stream>>>(
        (const float*)d_in[cv[i].src_idx], wbf + cv[i].off, cv[i].n);

  const int ntiles = (N + 15) / 16;

  init_kernel<<<(out_size + 255) / 256, 256, 0, stream>>>(
      (float*)d_out, out_size, mseg, zseg, B);

  att_branch_kernel<<<ntiles, 32, 0, stream>>>(
      h_sca, h_vec, temb, t_ptr, wbf, a1bg, a2Ws, att, N);

  segmax_kernel<<<(N + 255) / 256, 256, 0, stream>>>(att, bid, mseg, N);
  segsum_kernel<<<(N + 255) / 256, 256, 0, stream>>>(att, bid, mseg, zseg, N);

  net_branch_kernel<<<ntiles, 32, 0, stream>>>(
      h_sca, h_vec, pos, temb, t_ptr, wbf, n1bg, n2bg,
      att, bid, mseg, zseg,
      out_feat, out_vec, out_pos, N);
}
